// globalAttention_22325240005127
// MI455X (gfx1250) — matrix operand order repaired and checked by reading, not compiled, not run on hardware
//
#include <hip/hip_runtime.h>

// Problem constants (match the reference)
#define B_    4
#define T_    7
#define C_    64
#define H_    128
#define W_    128
#define P_    8
#define NP_   256        // patches per image = (128/8)^2
#define NTOK_ 1792       // T_ * NP_
#define D_    4096       // C_ * P_ * P_
#define HW_   (H_ * W_)
#define SCALE_ 0.015625f // d^-0.5 = 1/sqrt(4096)

typedef __attribute__((ext_vector_type(16))) _Float16 v16h;
typedef __attribute__((ext_vector_type(8)))  _Float16 v8h;
typedef __attribute__((ext_vector_type(8)))  float    v8f;

union AFrag { v16h v; v8h h[2]; };

#define WMMA_F16(a, b, c) \
  __builtin_amdgcn_wmma_f32_16x16x32_f16(false, (a), false, (b), (short)0, (c), false, false)

// ---------------------------------------------------------------------------
// WMMA fragment helpers (wave32; A/B use this code's contiguous K order, k = 16h + s)
// A: 16x32 f16. lane<16: M=lane, K=0..15;
//    lane>=16: M=lane-16, K=16..31
// ---------------------------------------------------------------------------
__device__ __forceinline__ v16h load_afrag(const _Float16* __restrict__ p, int ld,
                                           int m0, int k0, int lane) {
  int l  = lane & 15;
  int hi = lane >> 4;
  const _Float16* base = p + (size_t)(m0 + l) * ld + k0 + hi * 16;
  AFrag f;
  f.v = *(const v16h*)(base);          // K = k0+hi*16 .. +15
  return f.v;
}

// B: 32x16 f16 representing X^T where X is row-major [N][K].
// lane<16: N=lane, K=k0..k0+15 ; lane>=16: N=lane-16, K=k0+16..k0+31
__device__ __forceinline__ v16h load_bfrag(const _Float16* __restrict__ p, int ld,
                                           int n0, int k0, int lane) {
  int l  = lane & 15;
  int hi = lane >> 4;
  return *(const v16h*)(p + (size_t)(n0 + l) * ld + k0 + hi * 16);
}

// C/D: 16x16 f32. lane<16: N=lane, VGPR r -> M=r ; lane>=16: N=lane-16, M=8+r
__device__ __forceinline__ void store_cfrag(float* __restrict__ p, int ld,
                                            int m0, int n0, int lane, v8f acc) {
  int l  = lane & 15;
  int hi = lane >> 4;
  float* base = p + (size_t)(m0 + hi * 8) * ld + n0 + l;
#pragma unroll
  for (int r = 0; r < 8; ++r) base[(size_t)r * ld] = acc[r];
}

// ---------------------------------------------------------------------------
// Batched GEMM: C[M][N] = A[M][K] * B[N][K]^T   (f16 in, f32 out)
// block = 256 threads = 8 waves (2 M x 4 N), wave tile 32x32, block tile 64x128
// grid = (N/128, M/64, batch)
// ---------------------------------------------------------------------------
__global__ __launch_bounds__(256) void wmma_gemm_nt(
    const _Float16* __restrict__ A, int lda, size_t strideA,
    const _Float16* __restrict__ Bm, int ldb, size_t strideB,
    float* __restrict__ Cm, int ldc, size_t strideC,
    int Kdim) {
  const int lane = threadIdx.x & 31;
  const int wid  = threadIdx.x >> 5;
  const int waveM = wid >> 2;   // 0..1
  const int waveN = wid & 3;    // 0..3

  const int m_base = blockIdx.y * 64  + waveM * 32;
  const int n_base = blockIdx.x * 128 + waveN * 32;

  const _Float16* Ab = A  + (size_t)blockIdx.z * strideA;
  const _Float16* Bb = Bm + (size_t)blockIdx.z * strideB;
  float*          Cb = Cm + (size_t)blockIdx.z * strideC;

  v8f acc00 = {}, acc01 = {}, acc10 = {}, acc11 = {};

#pragma unroll 2
  for (int k0 = 0; k0 < Kdim; k0 += 32) {
    v16h a0 = load_afrag(Ab, lda, m_base,      k0, lane);
    v16h a1 = load_afrag(Ab, lda, m_base + 16, k0, lane);
    v16h b0 = load_bfrag(Bb, ldb, n_base,      k0, lane);
    v16h b1 = load_bfrag(Bb, ldb, n_base + 16, k0, lane);

    if (k0 + 32 < Kdim) {  // pull next K-slab toward the WGP (global_prefetch)
      __builtin_prefetch(Ab + (size_t)(m_base + (lane & 15)) * lda + k0 + 32, 0, 1);
      __builtin_prefetch(Bb + (size_t)(n_base + (lane & 15)) * ldb + k0 + 32, 0, 1);
    }

    acc00 = WMMA_F16(a0, b0, acc00);
    acc01 = WMMA_F16(a0, b1, acc01);
    acc10 = WMMA_F16(a1, b0, acc10);
    acc11 = WMMA_F16(a1, b1, acc11);
  }

  store_cfrag(Cb, ldc, m_base,      n_base,      lane, acc00);
  store_cfrag(Cb, ldc, m_base,      n_base + 16, lane, acc01);
  store_cfrag(Cb, ldc, m_base + 16, n_base,      lane, acc10);
  store_cfrag(Cb, ldc, m_base + 16, n_base + 16, lane, acc11);
}

// ---------------------------------------------------------------------------
// Implicit-GEMM 3x3 dense conv (64->64) building blocks.
// Input Xt[n][pix][ci] f16 (pixel-major), weights Wh[tap][co][ci] f16.
// B-fragment: lane = output pixel, 16 contiguous ci per lane-half; halo lanes
// return zero columns (D = A*0 + C leaves acc untouched -> correct zero-pad).
// ---------------------------------------------------------------------------
__device__ __forceinline__ v16h load_bfrag_pix(const _Float16* __restrict__ Xn,
                                               int p0, int k0, int lane,
                                               int dy, int dx) {
  int l  = lane & 15;
  int hi = lane >> 4;
  int p  = p0 + l;
  int yy = (p >> 7) + dy;
  int xx = (p & 127) + dx;
  if ((unsigned)yy >= (unsigned)H_ || (unsigned)xx >= (unsigned)W_) {
    v16h z = {};
    return z;
  }
  return *(const v16h*)(Xn + ((size_t)(yy * W_ + xx)) * C_ + k0 + hi * 16);
}

__device__ __forceinline__ void conv_accum(const _Float16* __restrict__ Xn,
                                           const _Float16* __restrict__ Wh,
                                           int m_base, int p_base, int lane,
                                           v8f& a00, v8f& a01, v8f& a10, v8f& a11) {
  for (int tap = 0; tap < 9; ++tap) {
    const int dy = tap / 3 - 1;
    const int dx = tap % 3 - 1;
    const _Float16* Wt = Wh + tap * C_ * C_;
#pragma unroll
    for (int k0 = 0; k0 < C_; k0 += 32) {
      v16h aA = load_afrag(Wt, C_, m_base,      k0, lane);
      v16h aB = load_afrag(Wt, C_, m_base + 16, k0, lane);
      v16h b0 = load_bfrag_pix(Xn, p_base,      k0, lane, dy, dx);
      v16h b1 = load_bfrag_pix(Xn, p_base + 16, k0, lane, dy, dx);
      a00 = WMMA_F16(aA, b0, a00);
      a01 = WMMA_F16(aA, b1, a01);
      a10 = WMMA_F16(aB, b0, a10);
      a11 = WMMA_F16(aB, b1, a11);
    }
  }
}

// conv_v: output tokenized straight into Vt[b][dd][tok] f16 (d-major == uv).
__device__ __forceinline__ void store_vtile(_Float16* __restrict__ Vt,
                                            const float* __restrict__ bv,
                                            int bb, int t, int m0, int p0,
                                            int lane, v8f acc) {
  const int l = lane & 15, hi = lane >> 4;
  const int p = p0 + l;
  const int yy = p >> 7, xx = p & 127;
  const int tok = t * NP_ + (yy >> 3) * 16 + (xx >> 3);
  const int pip = (yy & 7) * 8 + (xx & 7);
#pragma unroll
  for (int r = 0; r < 8; ++r) {
    const int co = m0 + hi * 8 + r;
    Vt[((size_t)bb * D_ + co * 64 + pip) * NTOK_ + tok] = (_Float16)(acc[r] + bv[co]);
  }
}

// grid = (HW/128, 1, B*T), block = 256 (8 waves: 2 co-waves x 4 pixel-waves)
__global__ __launch_bounds__(256) void convv_wmma(
    const _Float16* __restrict__ Xt, const _Float16* __restrict__ Whv,
    const float* __restrict__ bv, _Float16* __restrict__ Vt) {
  const int lane = threadIdx.x & 31;
  const int wid  = threadIdx.x >> 5;
  const int m_base = (wid >> 2) * 32;
  const int p_base = blockIdx.x * 128 + (wid & 3) * 32;
  const int n = blockIdx.z;
  const _Float16* Xn = Xt + (size_t)n * HW_ * C_;

  v8f a00 = {}, a01 = {}, a10 = {}, a11 = {};
  conv_accum(Xn, Whv, m_base, p_base, lane, a00, a01, a10, a11);

  const int bb = n / T_, t = n % T_;
  store_vtile(Vt, bv, bb, t, m_base,      p_base,      lane, a00);
  store_vtile(Vt, bv, bb, t, m_base,      p_base + 16, lane, a01);
  store_vtile(Vt, bv, bb, t, m_base + 16, p_base,      lane, a10);
  store_vtile(Vt, bv, bb, t, m_base + 16, p_base + 16, lane, a11);
}

// conv_c: bias + residual epilogue, fp32 output (coalesced: lane = pixel).
__device__ __forceinline__ void store_ctile(float* __restrict__ out,
                                            const float* __restrict__ xres,
                                            const float* __restrict__ bc,
                                            int n, int m0, int p0, int lane, v8f acc) {
  const int l = lane & 15, hi = lane >> 4;
  const int p = p0 + l;
#pragma unroll
  for (int r = 0; r < 8; ++r) {
    const int co = m0 + hi * 8 + r;
    const size_t idx = ((size_t)n * C_ + co) * HW_ + p;
    out[idx] = acc[r] + bc[co] + xres[idx];
  }
}

__global__ __launch_bounds__(256) void convc_wmma(
    const _Float16* __restrict__ Xc, const _Float16* __restrict__ Whc,
    const float* __restrict__ bc, const float* __restrict__ xres,
    float* __restrict__ out) {
  const int lane = threadIdx.x & 31;
  const int wid  = threadIdx.x >> 5;
  const int m_base = (wid >> 2) * 32;
  const int p_base = blockIdx.x * 128 + (wid & 3) * 32;
  const int n = blockIdx.z;
  const _Float16* Xn = Xc + (size_t)n * HW_ * C_;

  v8f a00 = {}, a01 = {}, a10 = {}, a11 = {};
  conv_accum(Xn, Whc, m_base, p_base, lane, a00, a01, a10, a11);

  store_ctile(out, xres, bc, n, m_base,      p_base,      lane, a00);
  store_ctile(out, xres, bc, n, m_base,      p_base + 16, lane, a01);
  store_ctile(out, xres, bc, n, m_base + 16, p_base,      lane, a10);
  store_ctile(out, xres, bc, n, m_base + 16, p_base + 16, lane, a11);
}

// ---------------------------------------------------------------------------
// Depthwise 3x3 conv for Q and K, fused with unfold/tokenize, output f16.
// Qh/Kh layout: [b][tok][dd], tok = t*256 + iq*16 + jq, dd = c*64 + pi*8 + pj.
// grid = (HW/256, C, B*T)
// ---------------------------------------------------------------------------
__global__ __launch_bounds__(256) void dwconv_qk_tok(
    const float* __restrict__ x,
    const float* __restrict__ wq, const float* __restrict__ bq,
    const float* __restrict__ wk, const float* __restrict__ bk,
    _Float16* __restrict__ Qh, _Float16* __restrict__ Kh) {
  const int n   = blockIdx.z;
  const int c   = blockIdx.y;
  const int pix = blockIdx.x * blockDim.x + threadIdx.x;
  const int yy  = pix >> 7;
  const int xx  = pix & 127;

  const float* xp = x + ((size_t)n * C_ + c) * HW_;
  float wq9[9], wk9[9];
#pragma unroll
  for (int i = 0; i < 9; ++i) { wq9[i] = wq[c * 9 + i]; wk9[i] = wk[c * 9 + i]; }

  float aq = bq[c];
  float ak = bk[c];
#pragma unroll
  for (int ki = 0; ki < 3; ++ki) {
    int sy = yy + ki - 1;
    if (sy < 0 || sy >= H_) continue;
#pragma unroll
    for (int kj = 0; kj < 3; ++kj) {
      int sx = xx + kj - 1;
      if (sx < 0 || sx >= W_) continue;
      float v = xp[sy * W_ + sx];
      aq = fmaf(wq9[ki * 3 + kj], v, aq);
      ak = fmaf(wk9[ki * 3 + kj], v, ak);
    }
  }

  const int bb = n / T_, t = n % T_;
  const int iq = yy >> 3, pi = yy & 7, jq = xx >> 3, pj = xx & 7;
  const int tok = t * NP_ + iq * 16 + jq;
  const int dd  = c * 64 + pi * 8 + pj;
  const size_t o = ((size_t)bb * NTOK_ + tok) * D_ + dd;
  Qh[o] = (_Float16)aq;
  Kh[o] = (_Float16)ak;
}

// ---------------------------------------------------------------------------
// Layout/packing helpers
// ---------------------------------------------------------------------------
// x (NCHW fp32) -> Xt[n][pix][ci] f16. grid = (HW/256, C, B*T)
__global__ __launch_bounds__(256) void xpose_f16(
    const float* __restrict__ x, _Float16* __restrict__ Xt) {
  const int n = blockIdx.z, ci = blockIdx.y;
  const int p = blockIdx.x * blockDim.x + threadIdx.x;
  Xt[((size_t)n * HW_ + p) * C_ + ci] =
      (_Float16)x[((size_t)n * C_ + ci) * HW_ + p];
}

// fold(O) -> Xc[n][pix][ci] f16 (pixel-major input for conv_c). Same grid.
__global__ __launch_bounds__(256) void fold_permute_f16(
    const float* __restrict__ O, _Float16* __restrict__ Xc) {
  const int n = blockIdx.z, ci = blockIdx.y;
  const int p = blockIdx.x * blockDim.x + threadIdx.x;
  const int bb = n / T_, t = n % T_;
  const int yy = p >> 7, xx = p & 127;
  const int tok = t * NP_ + (yy >> 3) * 16 + (xx >> 3);
  const int dd  = ci * 64 + (yy & 7) * 8 + (xx & 7);
  Xc[((size_t)n * HW_ + p) * C_ + ci] =
      (_Float16)O[((size_t)bb * NTOK_ + tok) * D_ + dd];
}

// w (O,I,3,3 fp32) -> Wh[tap][co][ci] f16. grid = (9*64*64/256)
__global__ __launch_bounds__(256) void w_tap_f16(
    const float* __restrict__ w, _Float16* __restrict__ Wh) {
  const int i = blockIdx.x * blockDim.x + threadIdx.x;
  if (i >= 9 * C_ * C_) return;
  const int tap = i / (C_ * C_);
  const int co  = (i / C_) % C_;
  const int ci  = i % C_;
  Wh[((size_t)tap * C_ + co) * C_ + ci] = (_Float16)w[((size_t)(co * C_ + ci)) * 9 + tap];
}

// ---------------------------------------------------------------------------
// Row softmax over S (len NTOK_ per row), scale folded in, f16 probs out.
// grid = (B*NTOK), block = 256
// ---------------------------------------------------------------------------
__global__ __launch_bounds__(256) void softmax_rows(
    const float* __restrict__ S, _Float16* __restrict__ Ph) {
  const int row = blockIdx.x;
  const float* r = S + (size_t)row * NTOK_;
  __shared__ float red[256];

  float mx = -1e30f;
  for (int i = threadIdx.x; i < NTOK_; i += 256)
    mx = fmaxf(mx, r[i] * SCALE_);
  red[threadIdx.x] = mx;
  __syncthreads();
  for (int s = 128; s > 0; s >>= 1) {
    if (threadIdx.x < s) red[threadIdx.x] = fmaxf(red[threadIdx.x], red[threadIdx.x + s]);
    __syncthreads();
  }
  mx = red[0];
  __syncthreads();

  float sum = 0.f;
  for (int i = threadIdx.x; i < NTOK_; i += 256)
    sum += __expf(r[i] * SCALE_ - mx);
  red[threadIdx.x] = sum;
  __syncthreads();
  for (int s = 128; s > 0; s >>= 1) {
    if (threadIdx.x < s) red[threadIdx.x] += red[threadIdx.x + s];
    __syncthreads();
  }
  const float inv = 1.0f / red[0];

  _Float16* outp = Ph + (size_t)row * NTOK_;
  for (int i = threadIdx.x; i < NTOK_; i += 256)
    outp[i] = (_Float16)(__expf(r[i] * SCALE_ - mx) * inv);
}

// ---------------------------------------------------------------------------
// Host launcher
// ---------------------------------------------------------------------------
extern "C" void kernel_launch(void* const* d_in, const int* in_sizes, int n_in,
                              void* d_out, int out_size, void* d_ws, size_t ws_size,
                              hipStream_t stream) {
  (void)in_sizes; (void)n_in; (void)out_size; (void)ws_size;

  const float* x  = (const float*)d_in[0];
  const float* wq = (const float*)d_in[1];
  const float* bq = (const float*)d_in[2];
  const float* wk = (const float*)d_in[3];
  const float* bk = (const float*)d_in[4];
  const float* wv = (const float*)d_in[5];
  const float* bv = (const float*)d_in[6];
  const float* wc = (const float*)d_in[7];
  const float* bc = (const float*)d_in[8];
  float* out = (float*)d_out;

  // Workspace layout (bytes, 256-aligned).
  const size_t szQK = (size_t)B_ * NTOK_ * D_ * sizeof(_Float16);       // 58,720,256
  const size_t szS  = (size_t)B_ * NTOK_ * NTOK_ * sizeof(float);       // 51,380,224
  const size_t szPh = (size_t)B_ * NTOK_ * NTOK_ * sizeof(_Float16);    // 25,690,112
  const size_t szX  = (size_t)B_ * T_ * HW_ * C_ * sizeof(_Float16);    // 58,720,256
  const size_t szW  = (size_t)9 * C_ * C_ * sizeof(_Float16);           // 73,728

  char* ws = (char*)d_ws;
  _Float16* Qh  = (_Float16*)(ws);
  _Float16* Kh  = (_Float16*)(ws + szQK);
  _Float16* Vt  = (_Float16*)(ws + 2 * szQK);
  float*    S   = (float*)   (ws + 3 * szQK);
  _Float16* Ph  = (_Float16*)(ws + 3 * szQK + szS);
  _Float16* Xt  = (_Float16*)(ws + 3 * szQK + szS + szPh);  // reused as Xc later
  _Float16* Whv = (_Float16*)(ws + 3 * szQK + szS + szPh + szX);
  _Float16* Whc = (_Float16*)(ws + 3 * szQK + szS + szPh + szX + szW);
  float*    O   = (float*)(ws);  // aliases Qh+Kh (dead after GEMM1); 2*szQK bytes

  const dim3 permGrid(HW_ / 256, C_, B_ * T_);
  const dim3 convGrid(HW_ / 128, 1, B_ * T_);

  // 0) pack weights to f16 tap-major
  w_tap_f16<<<dim3((9 * C_ * C_ + 255) / 256), 256, 0, stream>>>(wv, Whv);
  w_tap_f16<<<dim3((9 * C_ * C_ + 255) / 256), 256, 0, stream>>>(wc, Whc);

  // 1) x -> pixel-major f16
  xpose_f16<<<permGrid, 256, 0, stream>>>(x, Xt);

  // 2) depthwise Q,K + tokenize (f16)
  dwconv_qk_tok<<<permGrid, 256, 0, stream>>>(x, wq, bq, wk, bk, Qh, Kh);

  // 3) V = conv3x3(x) via WMMA implicit GEMM, tokenized d-major
  convv_wmma<<<convGrid, 256, 0, stream>>>(Xt, Whv, bv, Vt);

  // 4) S = Qh * Kh^T   (M=N=1792, K=4096)
  wmma_gemm_nt<<<dim3(NTOK_ / 128, NTOK_ / 64, B_), 256, 0, stream>>>(
      Qh, D_, (size_t)NTOK_ * D_,
      Kh, D_, (size_t)NTOK_ * D_,
      S, NTOK_, (size_t)NTOK_ * NTOK_,
      D_);

  // 5) softmax rows -> f16 probabilities
  softmax_rows<<<dim3(B_ * NTOK_), 256, 0, stream>>>(S, Ph);

  // 6) O = Ph * Vt^T   (M=1792, N=4096, K=1792); O overwrites Qh/Kh region
  wmma_gemm_nt<<<dim3(D_ / 128, NTOK_ / 64, B_), 256, 0, stream>>>(
      Ph, NTOK_, (size_t)NTOK_ * NTOK_,
      Vt, NTOK_, (size_t)D_ * NTOK_,
      O, D_, (size_t)NTOK_ * D_,
      NTOK_);

  // 7) fold(O) -> pixel-major f16 (reuse Xt buffer)
  fold_permute_f16<<<permGrid, 256, 0, stream>>>(O, Xt);

  // 8) out = conv3x3(fold(O)) + bias + residual, via WMMA implicit GEMM
  convc_wmma<<<convGrid, 256, 0, stream>>>(Xt, Whc, bc, x, out);
}
